// FalconH1SSMDecoderLayer_46273977647120
// MI455X (gfx1250) — compile-verified, matrix operand order checked by reading, not run on hardware
//
#include <hip/hip_runtime.h>
#include <hip/hip_bf16.h>
#include <cstdint>
#include <cstddef>

// ---------------------------------------------------------------------------
// FalconH1 SSM decoder layer for gfx1250 (MI455X).
//   GEMMs: v_wmma_f32_16x16x32_bf16 (f32 accum), double-buffered LDS tiles,
//          A staged with global_load_async_to_lds_b128 overlapped with WMMA.
//   Scan:  S=128 state split across 2 thread-halves, state in VGPRs.
// ---------------------------------------------------------------------------

typedef __attribute__((ext_vector_type(16))) __bf16 v16bf;
typedef __attribute__((ext_vector_type(8)))  float  v8f;

union Frag { uint4 u[2]; v16bf v; };

#define HID_    2048
#define DSSM_   4096
#define S_      128
#define H_      64
#define P_      64
#define CONVD_  4352
#define PROJ_   8512
#define LSEQ_   2048
#define BATCH_  2
#define MROWS_  4096   // BATCH_*LSEQ_

// ---------------- f32 -> bf16 conversion ----------------
__global__ void f32_to_bf16_kernel(const float* __restrict__ in,
                                   __bf16* __restrict__ out, size_t n) {
  size_t i = (size_t)blockIdx.x * blockDim.x + threadIdx.x;
  if (i < n) out[i] = (__bf16)in[i];
}

// ---------------- WMMA bf16 GEMM:  C(MxN) = A(MxK) * B(KxN) ----------------
// Block: 256 threads (8 waves). Block tile 256x64, K-step 64, double-buffered.
// Wave tile 32x64 -> 2 A-frags x 4 B-frags -> 16 WMMAs per K-step.
#define BM 256
#define BN 64
#define BK 64
#define LDS_STR 72   // 64 + 8 bf16 pad -> 144B rows: 16B aligned, conflict-free

__global__ __launch_bounds__(256)
void wmma_gemm_bf16(const __bf16* __restrict__ A, const __bf16* __restrict__ B,
                    float* __restrict__ C, const float* __restrict__ colScale,
                    int M, int N, int K) {
  (void)M;
  __shared__ __bf16 As[2][BM][LDS_STR];
  __shared__ __bf16 Bs[2][BN][LDS_STR];   // stored transposed: [n][k]

  const int tid  = threadIdx.x;
  const int lane = tid & 31;
  const int wave = tid >> 5;           // 0..7 -> 32-row strip within BM
  const int r    = lane & 15;
  const int hi   = lane >> 4;

  const int bm = blockIdx.y * BM;
  const int bn = blockIdx.x * BN;

  // staging coordinates (8-bf16 = 16B chunks)
  const int srow = tid >> 3;           // 0..31
  const int scol = (tid & 7) * 8;      // 0..56

  // per-buffer LDS byte offsets for async A staging
  const unsigned ldsA[2] = {
    (unsigned)(uintptr_t)(&As[0][0][0]) + (unsigned)(srow * LDS_STR + scol) * 2u,
    (unsigned)(uintptr_t)(&As[1][0][0]) + (unsigned)(srow * LDS_STR + scol) * 2u
  };

  // issue async copy of A tile (256x64) for k-offset k0 into buffer buf
  auto stageA = [&](int buf, int k0) {
    const __bf16* aBase = A + (size_t)(bm + srow) * K + k0 + scol;
#pragma unroll
    for (int i = 0; i < 8; ++i) {                 // rows srow + 32*i
      const unsigned lds = ldsA[buf] + (unsigned)(i * 32 * LDS_STR) * 2u;
      const __bf16* gp = aBase + (size_t)(i * 32) * K;
      asm volatile("global_load_async_to_lds_b128 %0, %1, off"
                   :: "v"(lds), "v"(gp) : "memory");
    }
  };
  // load B tile chunk (64x64) into VGPRs
  auto loadB = [&](uint4* bu, int k0) {
#pragma unroll
    for (int i = 0; i < 2; ++i)
      bu[i] = *(const uint4*)(B + (size_t)(k0 + srow + i * 32) * N + bn + scol);
  };
  // transpose-store B chunk into Bs[buf]  ([n][k])
  auto storeB = [&](int buf, const uint4* bu) {
#pragma unroll
    for (int i = 0; i < 2; ++i) {
      union { uint4 u; __bf16 h[8]; } t; t.u = bu[i];
#pragma unroll
      for (int j = 0; j < 8; ++j) Bs[buf][scol + j][srow + i * 32] = t.h[j];
    }
  };

  v8f acc[2][4] = {};

  // prologue: stage tile 0 into buffer 0
  stageA(0, 0);
  { uint4 b0[2]; loadB(b0, 0); storeB(0, b0); }

  int cur = 0;
  for (int k0 = 0; k0 < K; k0 += BK) {
    const bool hasNext = (k0 + BK < K);
    uint4 bnx[2];
    if (hasNext) loadB(bnx, k0 + BK);            // B global loads in flight

    asm volatile("s_wait_asynccnt 0x0" ::: "memory");  // A[cur] complete
    __syncthreads();

    if (hasNext) stageA(cur ^ 1, k0 + BK);       // DMA streams during compute
    if (k0 + 2 * BK < K)
      __builtin_prefetch(B + (size_t)(k0 + 2 * BK + srow) * N + bn + scol, 0, 3);

    // ---- compute from buffer cur: 2 K-substeps of 32, 8 WMMAs each ---------
    const __bf16 (*Ac)[LDS_STR] = As[cur];
    const __bf16 (*Bc)[LDS_STR] = Bs[cur];
    const int wrow = wave * 32;
#pragma unroll
    for (int ks = 0; ks < BK; ks += 32) {
      // A frags: lane row = wrow + mt*16 + r; elems 0..7 -> K=hi*8+0..7,
      // elems 8..15 -> K=16+hi*8+0..7 (ISA 16-bit 16x32 A layout)
      Frag fa0, fa1;
      fa0.u[0] = *(const uint4*)&Ac[wrow + r][ks + hi * 8];
      fa0.u[1] = *(const uint4*)&Ac[wrow + r][ks + 16 + hi * 8];
      fa1.u[0] = *(const uint4*)&Ac[wrow + 16 + r][ks + hi * 8];
      fa1.u[1] = *(const uint4*)&Ac[wrow + 16 + r][ks + 16 + hi * 8];
#pragma unroll
      for (int nt = 0; nt < 4; ++nt) {
        Frag fb;
        fb.u[0] = *(const uint4*)&Bc[nt * 16 + r][ks + hi * 8];
        fb.u[1] = *(const uint4*)&Bc[nt * 16 + r][ks + 16 + hi * 8];
        acc[0][nt] = __builtin_amdgcn_wmma_f32_16x16x32_bf16(
            false, fa0.v, false, fb.v, 0, acc[0][nt], false, false);
        acc[1][nt] = __builtin_amdgcn_wmma_f32_16x16x32_bf16(
            false, fa1.v, false, fb.v, 0, acc[1][nt], false, false);
      }
    }

    if (hasNext) storeB(cur ^ 1, bnx);           // after math: no fence in front
    __syncthreads();
    cur ^= 1;
  }

  // ---- epilogue: C/D layout -> VGPR i: row = hi*8 + i, col = lane&15 -------
#pragma unroll
  for (int nt = 0; nt < 4; ++nt) {
    const int col = bn + nt * 16 + r;
    const float s = colScale ? colScale[col] : 1.0f;
#pragma unroll
    for (int mt = 0; mt < 2; ++mt) {
#pragma unroll
      for (int i = 0; i < 8; ++i) {
        const int row = bm + wave * 32 + mt * 16 + hi * 8 + i;
        C[(size_t)row * N + col] = acc[mt][nt][i] * s;
      }
    }
  }
}

// ---------------- causal depthwise conv1d (K=4) + SiLU over xBC slice ------
__global__ void conv_silu_kernel(const float* __restrict__ proj,
                                 const float* __restrict__ conv_w,
                                 const float* __restrict__ conv_b,
                                 float* __restrict__ xc) {
  int idx = blockIdx.x * blockDim.x + threadIdx.x;  // MROWS_*CONVD_
  if (idx >= MROWS_ * CONVD_) return;
  const int c = idx % CONVD_;
  const int m = idx / CONVD_;
  const int t = m & (LSEQ_ - 1);
  const int b = m >> 11;
  float v = conv_b[c];
#pragma unroll
  for (int k = 0; k < 4; ++k) {
    const int tt = t + k - 3;
    if (tt >= 0)
      v += conv_w[c * 4 + k] *
           proj[(size_t)(b * LSEQ_ + tt) * PROJ_ + DSSM_ + c];
  }
  xc[(size_t)m * CONVD_ + c] = v / (1.0f + __expf(-v));  // SiLU
}

// ---------------- selective scan ---------------------------------------------
// Block per (h,b); 128 threads = (p:64) x (s-half:2); 64 states per thread in
// VGPRs. Halved dependent-FMA chain per step; y combined through LDS.
__global__ __launch_bounds__(128)
void ssm_scan_kernel(const float* __restrict__ proj, const float* __restrict__ xc,
                     const float* __restrict__ A_log, const float* __restrict__ Dp,
                     const float* __restrict__ dt_bias, float* __restrict__ y) {
  const int h    = blockIdx.x;
  const int b    = blockIdx.y;
  const int tid  = threadIdx.x;
  const int p    = tid & 63;
  const int half = tid >> 6;          // 0: s=0..63, 1: s=64..127
  const int sb   = half * 64;

  __shared__ float Bsh[S_];
  __shared__ float Csh[S_];
  __shared__ float Ysh[128];

  float st[64];
#pragma unroll
  for (int s = 0; s < 64; ++s) st[s] = 0.0f;

  const float negA = -__expf(A_log[h]);
  const float Dh   = Dp[h];
  const float dtb  = dt_bias[h];

  for (int t = 0; t < LSEQ_; ++t) {
    const size_t m  = (size_t)b * LSEQ_ + t;
    const float* xr = xc + m * CONVD_;

    Bsh[tid] = xr[DSSM_ + tid];          // 128 threads load all of B_t
    Csh[tid] = xr[DSSM_ + S_ + tid];     // and C_t

    const float xv  = xr[h * P_ + p];
    const float dtr = proj[m * PROJ_ + (PROJ_ - H_) + h] + dtb;
    const float dt  = (dtr > 20.0f) ? dtr : log1pf(__expf(dtr));  // softplus
    const float dA  = __expf(dt * negA);
    const float dtx = dt * xv;
    __syncthreads();

    float yv = 0.0f;
#pragma unroll
    for (int s = 0; s < 64; ++s) {
      st[s] = fmaf(st[s], dA, dtx * Bsh[sb + s]);
      yv    = fmaf(st[s], Csh[sb + s], yv);
    }
    Ysh[tid] = yv;
    __syncthreads();
    if (half == 0)
      y[m * DSSM_ + h * P_ + p] = fmaf(Dh, xv, Ysh[p] + Ysh[p + 64]);
  }
}

// ---------------- gated RMSNorm, emit bf16 for out_proj ---------------------
__global__ __launch_bounds__(256)
void gated_rmsnorm_bf16(const float* __restrict__ proj, const float* __restrict__ y,
                        const float* __restrict__ norm_w, __bf16* __restrict__ g) {
  const int m = blockIdx.x;
  const int tid = threadIdx.x;
  __shared__ float red[256];

  float loc[16];
  float ss = 0.0f;
#pragma unroll
  for (int j = 0; j < 16; ++j) {
    const int idx = j * 256 + tid;
    const float z  = proj[(size_t)m * PROJ_ + idx];       // z block
    const float yv = y[(size_t)m * DSSM_ + idx];
    const float gv = yv * (z / (1.0f + __expf(-z)));      // y * silu(z)
    loc[j] = gv;
    ss += gv * gv;
  }
  red[tid] = ss;
  __syncthreads();
  for (int o = 128; o > 0; o >>= 1) {
    if (tid < o) red[tid] += red[tid + o];
    __syncthreads();
  }
  const float rr = rsqrtf(red[0] * (1.0f / DSSM_) + 1e-5f);
#pragma unroll
  for (int j = 0; j < 16; ++j) {
    const int idx = j * 256 + tid;
    g[(size_t)m * DSSM_ + idx] = (__bf16)(loc[j] * rr * norm_w[idx]);
  }
}

// ---------------- flat f32 copy (residual passthrough) ----------------------
__global__ void copy_f32_kernel(const float* __restrict__ in,
                                float* __restrict__ out, size_t n) {
  size_t i = (size_t)blockIdx.x * blockDim.x + threadIdx.x;
  if (i < n) out[i] = in[i];
}

// ---------------------------------------------------------------------------
static inline size_t alignup(size_t x) { return (x + 255) & ~(size_t)255; }

extern "C" void kernel_launch(void* const* d_in, const int* in_sizes, int n_in,
                              void* d_out, int out_size, void* d_ws, size_t ws_size,
                              hipStream_t stream) {
  (void)in_sizes; (void)n_in; (void)out_size; (void)ws_size;

  const float* hidden   = (const float*)d_in[0];   // (B,L,HID)
  const float* residual = (const float*)d_in[1];   // (B,L,HID)
  const float* W_in     = (const float*)d_in[2];   // (HID,PROJ)
  const float* conv_w   = (const float*)d_in[3];   // (CONVD,4)
  const float* conv_b   = (const float*)d_in[4];   // (CONVD,)
  const float* A_log    = (const float*)d_in[5];   // (H,)
  const float* Dp       = (const float*)d_in[6];   // (H,)
  const float* dt_bias  = (const float*)d_in[7];   // (H,)
  const float* norm_w   = (const float*)d_in[8];   // (DSSM,)
  const float* W_out    = (const float*)d_in[9];   // (DSSM,HID)
  const float* mup      = (const float*)d_in[10];  // (PROJ,)

  float* out = (float*)d_out;                      // [out (B,L,HID)][residual]

  // workspace carve-up
  char* w = (char*)d_ws;
  float* proj = (float*)w;                w += alignup((size_t)MROWS_ * PROJ_ * 4);
  float* xc   = (float*)w;                w += alignup((size_t)MROWS_ * CONVD_ * 4);
  float* ybuf = (float*)w;                w += alignup((size_t)MROWS_ * DSSM_ * 4);
  __bf16* hbf  = (__bf16*)w;              w += alignup((size_t)MROWS_ * HID_ * 2);
  __bf16* wibf = (__bf16*)w;              w += alignup((size_t)HID_ * PROJ_ * 2);
  __bf16* wobf = (__bf16*)w;              w += alignup((size_t)DSSM_ * HID_ * 2);
  __bf16* gbf  = (__bf16*)w;              w += alignup((size_t)MROWS_ * DSSM_ * 2);

  const size_t nHid  = (size_t)MROWS_ * HID_;      // 8388608
  const size_t nWin  = (size_t)HID_ * PROJ_;       // 17432576
  const size_t nWout = (size_t)DSSM_ * HID_;       // 8388608

  // 1) one-time bf16 casts (bandwidth-trivial)
  f32_to_bf16_kernel<<<(int)((nHid  + 255) / 256), 256, 0, stream>>>(hidden, hbf,  nHid);
  f32_to_bf16_kernel<<<(int)((nWin  + 255) / 256), 256, 0, stream>>>(W_in,   wibf, nWin);
  f32_to_bf16_kernel<<<(int)((nWout + 255) / 256), 256, 0, stream>>>(W_out,  wobf, nWout);

  // 2) in_proj GEMM + fused mup scaling: proj = (hidden @ W_in) * mup
  {
    dim3 grid(PROJ_ / BN, MROWS_ / BM);  // (133, 16)
    wmma_gemm_bf16<<<grid, 256, 0, stream>>>(hbf, wibf, proj, mup,
                                             MROWS_, PROJ_, HID_);
  }

  // 3) causal depthwise conv1d + SiLU over xBC
  {
    const int n = MROWS_ * CONVD_;
    conv_silu_kernel<<<(n + 255) / 256, 256, 0, stream>>>(proj, conv_w, conv_b, xc);
  }

  // 4) selective scan
  {
    dim3 grid(H_, BATCH_);
    ssm_scan_kernel<<<grid, 128, 0, stream>>>(proj, xc, A_log, Dp, dt_bias, ybuf);
  }

  // 5) gated RMSNorm -> bf16
  gated_rmsnorm_bf16<<<MROWS_, 256, 0, stream>>>(proj, ybuf, norm_w, gbf);

  // 6) out_proj GEMM into d_out
  {
    dim3 grid(HID_ / BN, MROWS_ / BM);   // (32, 16)
    wmma_gemm_bf16<<<grid, 256, 0, stream>>>(gbf, wobf, out, nullptr,
                                             MROWS_, HID_, DSSM_);
  }

  // 7) residual passthrough (second tuple element)
  copy_f32_kernel<<<(int)((nHid + 255) / 256), 256, 0, stream>>>(
      residual, out + nHid, nHid);
}
